// Lex3d_61108794687740
// MI455X (gfx1250) — matrix operand order repaired and checked by reading, not compiled, not run on hardware
//
#include <hip/hip_runtime.h>
#include <hip/hip_bf16.h>
#include <stdint.h>

// ---------------- problem constants ----------------
constexpr int N_NODES = 100000;
constexpr int N_EDGES = 500000;
constexpr int DIM     = 256;   // D
constexpr int HDIM    = 128;   // H
constexpr int DEDIM   = 64;    // DE
constexpr int P3      = 3 * HDIM; // 384

typedef __attribute__((ext_vector_type(16))) __bf16 v16bf;
typedef __attribute__((ext_vector_type(8)))  float  v8f;

union FragBF { v16bf v; uint4 q[2]; };

__device__ __forceinline__ unsigned short f2bf(float f) {
  unsigned u = __float_as_uint(f);
  u += 0x7FFFu + ((u >> 16) & 1u);          // round-to-nearest-even
  return (unsigned short)(u >> 16);
}
__device__ __forceinline__ float eluf(float v)   { return v > 0.f ? v : expm1f(v); }
__device__ __forceinline__ float lreluf(float v) { return v > 0.f ? v : 0.01f * v; }
// order-preserving float<->uint encode for atomicMax on signed floats
__device__ __forceinline__ unsigned fenc(float f) {
  unsigned u = __float_as_uint(f);
  return (u & 0x80000000u) ? ~u : (u | 0x80000000u);
}
__device__ __forceinline__ float fdec(unsigned e) {
  return (e & 0x80000000u) ? __uint_as_float(e & 0x7FFFFFFFu) : __uint_as_float(~e);
}

// =====================================================================
// WMMA GEMM: C[M,Nc] = A[M,K](bf16 rowmaj) * B (given as Bt[Nc,K] bf16)
//            optional +bias[col], optional +=C (ACC), optional A-row gather.
// 8 waves/block; wave computes 16 rows x 128 cols (8 wmma tiles sharing
// one A fragment -> halves HBM passes over the big E-row A matrices).
// Requires Nc % 128 == 0 (all Nc here: 384, 128, 256).
// =====================================================================
template<bool GATHER, bool ACC>
__global__ __launch_bounds__(256) void wmma_gemm(
    const unsigned short* __restrict__ A, const unsigned short* __restrict__ Bt,
    const float* __restrict__ bias, float* __restrict__ C,
    const int* __restrict__ rowidx, int M, int Nc, int K, int lda)
{
  const int lane    = threadIdx.x & 31;
  const int wave    = threadIdx.x >> 5;
  const int rowBase = blockIdx.y * 128 + wave * 16;
  const int colBase = blockIdx.x * 128;
  const int lrow = lane & 15;
  const int hi   = lane >> 4;        // lane group 0/1
  const int offA = hi ? 8 : 0;       // bf16 A frag: K = offA + {0..7, 16..23}
  const int offB = hi ? 8 : 0;       // bf16 B frag: K = offB + {0..7, 16..23}

  int rA  = rowBase + lrow;
  int rCl = rA < M ? rA : (M - 1);
  int64_t aRow = GATHER ? (int64_t)rowidx[rCl] : (int64_t)rCl;
  const unsigned short* Ap = A + aRow * (int64_t)lda;

  const int rowS = rowBase + 8 * hi; // C frag: M = rowS + r

  v8f acc[8];
#pragma unroll
  for (int n = 0; n < 8; ++n)
#pragma unroll
    for (int r = 0; r < 8; ++r) acc[n][r] = 0.f;

  if constexpr (ACC) {
#pragma unroll
    for (int n = 0; n < 8; ++n) {
      int c = colBase + 16 * n + lrow;
#pragma unroll
      for (int r = 0; r < 8; ++r) {
        int rr = rowS + r;
        if (rr < M) acc[n][r] = C[(size_t)rr * Nc + c];
      }
    }
  }

  for (int kk = 0; kk < K; kk += 32) {
    FragBF af;
    af.q[0] = *(const uint4*)(Ap + kk + offA);
    af.q[1] = *(const uint4*)(Ap + kk + offA + 16);
    if (kk + 32 < K) __builtin_prefetch(Ap + kk + 32, 0, 1);
#pragma unroll
    for (int n = 0; n < 8; ++n) {
      const unsigned short* Bp = Bt + (size_t)(colBase + 16 * n + lrow) * K + kk + offB;
      FragBF bfg;
      bfg.q[0] = *(const uint4*)(Bp);
      bfg.q[1] = *(const uint4*)(Bp + 16);
      acc[n] = __builtin_amdgcn_wmma_f32_16x16x32_bf16(
          false, af.v, false, bfg.v, (short)0, acc[n], false, false);
    }
  }

#pragma unroll
  for (int n = 0; n < 8; ++n) {
    int c  = colBase + 16 * n + lrow;
    float bv = bias ? bias[c] : 0.f;
#pragma unroll
    for (int r = 0; r < 8; ++r) {
      int rr = rowS + r;
      if (rr < M) C[(size_t)rr * Nc + c] = acc[n][r] + bv;
    }
  }
}

// message GEMM: Out[dst[row], :] += (A@B)[row, :] * ascale[row]  (atomic scatter)
__global__ __launch_bounds__(256) void wmma_gemm_scatter(
    const unsigned short* __restrict__ A, const unsigned short* __restrict__ Bt,
    const float* __restrict__ ascale, const int* __restrict__ dst,
    float* __restrict__ Out, int M, int Nc, int K, int lda)
{
  const int lane    = threadIdx.x & 31;
  const int wave    = threadIdx.x >> 5;
  const int rowBase = blockIdx.y * 128 + wave * 16;
  const int colBase = blockIdx.x * 128;
  const int lrow = lane & 15;
  const int hi   = lane >> 4;
  const int offA = hi ? 8 : 0;
  const int offB = hi ? 8 : 0;

  int rA  = rowBase + lrow;
  int rCl = rA < M ? rA : (M - 1);
  const unsigned short* Ap = A + (int64_t)rCl * lda;
  const int rowS = rowBase + 8 * hi;

  v8f acc[8];
#pragma unroll
  for (int n = 0; n < 8; ++n)
#pragma unroll
    for (int r = 0; r < 8; ++r) acc[n][r] = 0.f;

  for (int kk = 0; kk < K; kk += 32) {
    FragBF af;
    af.q[0] = *(const uint4*)(Ap + kk + offA);
    af.q[1] = *(const uint4*)(Ap + kk + offA + 16);
#pragma unroll
    for (int n = 0; n < 8; ++n) {
      const unsigned short* Bp = Bt + (size_t)(colBase + 16 * n + lrow) * K + kk + offB;
      FragBF bfg;
      bfg.q[0] = *(const uint4*)(Bp);
      bfg.q[1] = *(const uint4*)(Bp + 16);
      acc[n] = __builtin_amdgcn_wmma_f32_16x16x32_bf16(
          false, af.v, false, bfg.v, (short)0, acc[n], false, false);
    }
  }

#pragma unroll
  for (int n = 0; n < 8; ++n) {
    int c = colBase + 16 * n + lrow;
#pragma unroll
    for (int r = 0; r < 8; ++r) {
      int rr = rowS + r;
      if (rr < M) {
        float s = ascale[rr];
        atomicAdd(&Out[(size_t)dst[rr] * Nc + c], acc[n][r] * s);
      }
    }
  }
}

// =====================================================================
// elementwise / reduction kernels
// =====================================================================
__global__ void k_fill0(float* p, size_t n) {
  size_t i = (size_t)blockIdx.x * blockDim.x + threadIdx.x;
  if (i < n) p[i] = 0.f;
}
__global__ void k_conv_bf16(const float* in, unsigned short* out, size_t n) {
  size_t i = (size_t)blockIdx.x * blockDim.x + threadIdx.x;
  if (i < n) out[i] = f2bf(in[i]);
}
// w[K][Ncol] fp32 -> wt[Ncol][K] bf16 (transposed weight staging)
__global__ void k_conv_wt(const float* w, unsigned short* wt, int K, int Ncol) {
  int i = blockIdx.x * blockDim.x + threadIdx.x;
  if (i < K * Ncol) {
    int k = i / Ncol, c = i % Ncol;
    wt[(size_t)c * K + k] = f2bf(w[i]);
  }
}
__global__ void k_concat3(const float* a, const float* b, const float* c, float* o) {
  int i = threadIdx.x + blockIdx.x * blockDim.x;
  if (i < HDIM) { o[i] = a[i]; o[HDIM + i] = b[i]; o[2 * HDIM + i] = c[i]; }
}
// row-wise: out = bf16( LN( ELU(in) ) ), one 256-thread block per row (D=256)
__global__ void k_elu_ln(const float* __restrict__ in, unsigned short* __restrict__ out) {
  __shared__ float red[DIM];
  int row = blockIdx.x, c = threadIdx.x;
  float v = eluf(in[(size_t)row * DIM + c]);
  red[c] = v; __syncthreads();
  for (int s = DIM / 2; s > 0; s >>= 1) { if (c < s) red[c] += red[c + s]; __syncthreads(); }
  float mu = red[0] * (1.f / DIM); __syncthreads();
  float d = v - mu; red[c] = d * d; __syncthreads();
  for (int s = DIM / 2; s > 0; s >>= 1) { if (c < s) red[c] += red[c + s]; __syncthreads(); }
  float inv = rsqrtf(red[0] * (1.f / DIM) + 1e-5f);
  out[(size_t)row * DIM + c] = f2bf((v - mu) * inv);
}
// h = elu(t + res (+ bias[col])) ; also refresh bf16 mirror
__global__ void k_resid_elu(const float* __restrict__ t, const float* __restrict__ res,
                            const float* __restrict__ bias, float* __restrict__ h,
                            unsigned short* __restrict__ hb, size_t n) {
  size_t i = (size_t)blockIdx.x * blockDim.x + threadIdx.x;
  if (i >= n) return;
  float v = t[i] + res[i] + (bias ? bias[i % DIM] : 0.f);
  v = eluf(v);
  h[i] = v; hb[i] = f2bf(v);
}
// xijk = bf16( elu( p[i0,0:128] + p[i1,128:256] + p[i2,256:384] + pe[e,:] ) )
__global__ void k_gather_combine(const float* __restrict__ p, const float* __restrict__ pe,
    const int* __restrict__ i0, const int* __restrict__ i1, const int* __restrict__ i2,
    unsigned short* __restrict__ out) {
  size_t t = (size_t)blockIdx.x * blockDim.x + threadIdx.x;
  if (t >= (size_t)N_EDGES * HDIM) return;
  int e = (int)(t / HDIM), c = (int)(t % HDIM);
  float v = p[(size_t)i0[e] * P3 + c] + p[(size_t)i1[e] * P3 + HDIM + c]
          + p[(size_t)i2[e] * P3 + 2 * HDIM + c] + pe[t];
  out[t] = f2bf(eluf(v));
}
// a[e] = lrelu( sum_c lrelu(s[e,c]) * att[c] ) ; one wave32 per edge
__global__ void k_att_score(const float* __restrict__ s, const float* __restrict__ att,
                            float* __restrict__ a) {
  int e = blockIdx.x * 8 + (threadIdx.x >> 5);
  int lane = threadIdx.x & 31;
  if (e >= N_EDGES) return;
  const float* row = s + (size_t)e * HDIM;
  float acc = 0.f;
  for (int c = lane; c < HDIM; c += 32) acc += lreluf(row[c]) * att[c];
  for (int off = 16; off > 0; off >>= 1) acc += __shfl_down(acc, off, 32);
  if (lane == 0) a[e] = lreluf(acc);
}
__global__ void k_segmax(const float* a, const int* dst, unsigned* m) {
  int e = blockIdx.x * blockDim.x + threadIdx.x;
  if (e < N_EDGES) atomicMax(&m[dst[e]], fenc(a[e]));
}
__global__ void k_expsum(const float* a, const int* dst, const unsigned* m,
                         float* ex, float* den) {
  int e = blockIdx.x * blockDim.x + threadIdx.x;
  if (e >= N_EDGES) return;
  float v = expf(a[e] - fdec(m[dst[e]]));
  ex[e] = v;
  atomicAdd(&den[dst[e]], v);
}
__global__ void k_rescale(const float* ex, const float* den, const float* hw,
                          const int* dst, float* a2, float* sums) {
  int e = blockIdx.x * blockDim.x + threadIdx.x;
  if (e >= N_EDGES) return;
  float v = ex[e] / (den[dst[e]] + 1e-16f) * hw[e];
  a2[e] = v;
  atomicAdd(&sums[dst[e]], v);
}
__global__ void k_renorm(float* a2, const float* sums, const int* dst) {
  int e = blockIdx.x * blockDim.x + threadIdx.x;
  if (e < N_EDGES) a2[e] = a2[e] / (sums[dst[e]] + 1e-5f);
}

// =====================================================================
// host orchestration
// =====================================================================
static void gemm(hipStream_t st, const unsigned short* A, const unsigned short* Bt,
                 const float* bias, float* C, const int* gidx, bool acc,
                 int M, int Nc, int K, int lda) {
  dim3 g(Nc / 128, (M + 127) / 128), b(256);
  if (gidx) {
    if (acc) wmma_gemm<true,  true ><<<g, b, 0, st>>>(A, Bt, bias, C, gidx, M, Nc, K, lda);
    else     wmma_gemm<true,  false><<<g, b, 0, st>>>(A, Bt, bias, C, gidx, M, Nc, K, lda);
  } else {
    if (acc) wmma_gemm<false, true ><<<g, b, 0, st>>>(A, Bt, bias, C, nullptr, M, Nc, K, lda);
    else     wmma_gemm<false, false><<<g, b, 0, st>>>(A, Bt, bias, C, nullptr, M, Nc, K, lda);
  }
}

extern "C" void kernel_launch(void* const* d_in, const int* in_sizes, int n_in,
                              void* d_out, int out_size, void* d_ws, size_t ws_size,
                              hipStream_t stream) {
  (void)in_sizes; (void)n_in; (void)out_size; (void)ws_size;
  auto F = [&](int i) { return (const float*)d_in[i]; };
  const float* X     = F(0);
  const float* EATTR = F(1);
  const float* HW    = F(2);
  const int*   HEIDX = (const int*)d_in[34];
  const int* i0 = HEIDX;
  const int* i1 = HEIDX + N_EDGES;
  const int* i2 = HEIDX + 2 * (size_t)N_EDGES;

  // ---- workspace carve-up ----
  unsigned char* W = (unsigned char*)d_ws;
  size_t off = 0;
  auto alloc = [&](size_t bytes) { void* p = W + off; off += (bytes + 255) & ~(size_t)255; return p; };
  float*          h    = (float*)alloc((size_t)N_NODES * DIM * 4);
  unsigned short* hb   = (unsigned short*)alloc((size_t)N_NODES * DIM * 2);
  float*          tmp  = (float*)alloc((size_t)N_NODES * DIM * 4);   // GEMM temp / agg accumulator
  unsigned short* ab   = (unsigned short*)alloc((size_t)N_NODES * DIM * 2); // LN output (bf16)
  unsigned short* xb   = (unsigned short*)alloc((size_t)N_NODES * DIM * 2); // x in bf16
  float*          p    = (float*)alloc((size_t)N_NODES * P3 * 4);    // vertex projections [N,384]
  float*          peS  = (float*)alloc((size_t)N_EDGES * HDIM * 4);  // pe, then s buffer
  unsigned short* xijk = (unsigned short*)alloc((size_t)N_EDGES * HDIM * 2);
  unsigned short* eb16 = (unsigned short*)alloc((size_t)N_EDGES * DEDIM * 2);
  unsigned short* w0   = (unsigned short*)alloc(512 * 1024);
  unsigned short* w1   = (unsigned short*)alloc(512 * 1024);
  float*          bcat = (float*)alloc(P3 * 4);
  float*          aE   = (float*)alloc((size_t)N_EDGES * 4);
  float*          exE  = (float*)alloc((size_t)N_EDGES * 4);
  float*          a2E  = (float*)alloc((size_t)N_EDGES * 4);
  unsigned*       mEnc = (unsigned*)alloc((size_t)N_NODES * 4);
  float*          den  = (float*)alloc((size_t)N_NODES * 4);
  float*          sums = (float*)alloc((size_t)N_NODES * 4);

  const size_t ND = (size_t)N_NODES * DIM;
  const size_t EH = (size_t)N_EDGES * HDIM;
  const int   EB  = (N_EDGES + 255) / 256;
  auto blk = [](size_t n) { return (unsigned)((n + 255) / 256); };

  // ---- one-time conversions ----
  k_conv_bf16<<<blk(ND), 256, 0, stream>>>(X, xb, ND);
  k_conv_bf16<<<blk((size_t)N_EDGES * DEDIM), 256, 0, stream>>>(EATTR, eb16, (size_t)N_EDGES * DEDIM);

  // ---- _transfer ----
  auto transfer = [&](const unsigned short* Ain, const float* res,
                      const float* wA, const float* wB) {
    k_conv_wt<<<blk(DIM * DIM), 256, 0, stream>>>(wA, w0, DIM, DIM);
    gemm(stream, Ain, w0, nullptr, tmp, nullptr, false, N_NODES, DIM, DIM, DIM);
    k_elu_ln<<<N_NODES, DIM, 0, stream>>>(tmp, ab);
    k_conv_wt<<<blk(DIM * DIM), 256, 0, stream>>>(wB, w1, DIM, DIM);
    gemm(stream, ab, w1, nullptr, tmp, nullptr, false, N_NODES, DIM, DIM, DIM);
    k_resid_elu<<<blk(ND), 256, 0, stream>>>(tmp, res, nullptr, h, hb, ND);
  };

  // ---- _tri ----
  auto tri = [&](int b, int uv) {
    const float *wi = F(b+0), *bi = F(b+1), *wj = F(b+2), *bj = F(b+3);
    const float *wk = F(b+4), *bk = F(b+5), *we = F(b+6), *be = F(b+7);
    const float *w1l = F(b+8), *w1r = F(b+9), *att = F(b+10), *w2 = F(b+11), *bias = F(b+12);
    const int* dst = HEIDX + (size_t)uv * N_EDGES;

    // fused vertex projections: p = hb @ [wi|wj|wk] + [bi|bj|bk]   (N x 384)
    k_conv_wt<<<blk(DIM * HDIM), 256, 0, stream>>>(wi, w0, DIM, HDIM);
    k_conv_wt<<<blk(DIM * HDIM), 256, 0, stream>>>(wj, w0 + (size_t)HDIM * DIM, DIM, HDIM);
    k_conv_wt<<<blk(DIM * HDIM), 256, 0, stream>>>(wk, w0 + (size_t)2 * HDIM * DIM, DIM, HDIM);
    k_concat3<<<1, HDIM, 0, stream>>>(bi, bj, bk, bcat);
    gemm(stream, hb, w0, bcat, p, nullptr, false, N_NODES, P3, DIM, DIM);

    // edge projection: pe = eattr @ we + be   (E x 128)
    k_conv_wt<<<blk(DEDIM * HDIM), 256, 0, stream>>>(we, w1, DEDIM, HDIM);
    gemm(stream, eb16, w1, be, peS, nullptr, false, N_EDGES, HDIM, DEDIM, DEDIM);

    // xijk = elu(gather-combine)  -> bf16
    k_gather_combine<<<blk(EH), 256, 0, stream>>>(p, peS, i0, i1, i2, xijk);

    // s = xijk @ w1l + h[dst] @ w1r   (E x 128, second GEMM gathers A rows + accumulates)
    k_conv_wt<<<blk(HDIM * HDIM), 256, 0, stream>>>(w1l, w0, HDIM, HDIM);
    gemm(stream, xijk, w0, nullptr, peS, nullptr, false, N_EDGES, HDIM, HDIM, HDIM);
    k_conv_wt<<<blk(DIM * HDIM), 256, 0, stream>>>(w1r, w1, DIM, HDIM);
    gemm(stream, hb, w1, nullptr, peS, dst, true, N_EDGES, HDIM, DIM, DIM);

    // attention logits
    k_att_score<<<(N_EDGES + 7) / 8, 256, 0, stream>>>(peS, att, aE);

    // segment softmax + edge-weight renorm
    k_fill0<<<blk(N_NODES), 256, 0, stream>>>((float*)mEnc, N_NODES);
    k_fill0<<<blk(N_NODES), 256, 0, stream>>>(den,  N_NODES);
    k_fill0<<<blk(N_NODES), 256, 0, stream>>>(sums, N_NODES);
    k_segmax <<<EB, 256, 0, stream>>>(aE, dst, mEnc);
    k_expsum <<<EB, 256, 0, stream>>>(aE, dst, mEnc, exE, den);
    k_rescale<<<EB, 256, 0, stream>>>(exE, den, HW, dst, a2E, sums);
    k_renorm <<<EB, 256, 0, stream>>>(a2E, sums, dst);

    // agg[dst] += (xijk @ w2) * a   (atomic scatter epilogue)
    k_fill0<<<blk(ND), 256, 0, stream>>>(tmp, ND);
    k_conv_wt<<<blk(HDIM * DIM), 256, 0, stream>>>(w2, w0, HDIM, DIM);
    dim3 g(DIM / 128, (N_EDGES + 127) / 128);
    wmma_gemm_scatter<<<g, 256, 0, stream>>>(xijk, w0, a2E, dst, tmp,
                                             N_EDGES, DIM, HDIM, HDIM);
    // h = elu(h + agg + bias)
    k_resid_elu<<<blk(ND), 256, 0, stream>>>(tmp, h, bias, h, hb, ND);
  };

  // ---- full network ----
  transfer(xb, X, F(3), F(4));      // transfer0 (residual = original x)
  tri(8, 2);                        // outward: update vertex k
  tri(21, 0);                       // inward: update vertex i
  transfer(hb, h, F(5), F(6));      // transfer1 (residual = current h)

  // exit: d_out = h @ w_exit
  k_conv_wt<<<blk(DIM * DIM), 256, 0, stream>>>(F(7), w0, DIM, DIM);
  gemm(stream, hb, w0, nullptr, (float*)d_out, nullptr, false, N_NODES, DIM, DIM, DIM);
}
